// FusionAttention_MMCA_sty_1606317769298
// MI455X (gfx1250) — matrix operand order repaired and checked by reading, not compiled, not run on hardware
//
#include <hip/hip_runtime.h>

#define TWO_N 8192
#define NMASK 4095
#define DD 512

typedef __attribute__((ext_vector_type(16))) __bf16 v16bf;
typedef __attribute__((ext_vector_type(8)))  float  v8f;

union V16U { v16bf v; uint4 u[2]; __bf16 e[16]; };

// A operand (16x32 bf16, row-major source, row = lane%16):
// half h=0: elems 0..15 -> K 0..15
// half h=1: elems 0..15 -> K 16..31
__device__ __forceinline__ v16bf load_a16(const __bf16* p, int h) {
  V16U r;
  r.u[0] = *(const uint4*)(p + 16 * h);
  r.u[1] = *(const uint4*)(p + 16 * h + 8);
  return r.v;
}
// B operand (32x16 bf16): col = lane%16, elem i -> K = 16*h + i.
// caller passes p already offset by +16*h; 16 consecutive bf16.
__device__ __forceinline__ v16bf load_b16(const __bf16* p) {
  V16U r;
  r.u[0] = *(const uint4*)(p);
  r.u[1] = *(const uint4*)(p + 8);
  return r.v;
}
__device__ __forceinline__ v8f wmma_bf16(v16bf a, v16bf b, v8f c) {
  return __builtin_amdgcn_wmma_f32_16x16x32_bf16(false, a, false, b,
                                                 (short)0, c, false, false);
}

// ---------------- prep: Fcat fp32 -> bf16 ----------------
__global__ void k_fcat(const float* __restrict__ R, const float* __restrict__ L,
                       __bf16* __restrict__ out) {
  size_t i = (size_t)blockIdx.x * 256 + threadIdx.x;  // 8192*512 total
  const size_t half = (size_t)4096 * 512;
  float v = (i < half) ? R[i] : L[i - half];
  out[i] = (__bf16)v;
}

// ---------------- prep: W fp32 [k][n] -> Wt bf16 [n][k] ----------------
__global__ void k_wt(const float* __restrict__ W, __bf16* __restrict__ Wt) {
  int i = blockIdx.x * 256 + threadIdx.x;  // 512*512 total
  int k = i >> 9, n = i & 511;
  Wt[(size_t)n * 512 + k] = (__bf16)W[(size_t)k * 512 + n];
}

// ---------------- prep: Er = (1/sqrt(D)) * [Distance ? sigmoid(exp(-(De/10)^2)) : 1]
__global__ void k_er(const float* __restrict__ De, const int* __restrict__ Dist,
                     float* __restrict__ Er) {
  size_t i = (size_t)blockIdx.x * 256 + threadIdx.x;  // 4096*4096 total
  float s = 0.044194173824159216f;  // 1/sqrt(512)
  if (*Dist) {
    float d = De[i];
    float g = __expf(-(d * 0.1f) * (d * 0.1f));
    s *= 1.0f / (1.0f + __expf(-g));
  }
  Er[i] = s;
}

// ---------------- generic 16x16-tile WMMA GEMM: X[rows x 512] @ Wt^T ----------------
// mode 0: bf16 row-major out; mode 1: bf16 transposed out (ld = rows)
// mode 2: bias+relu, bf16 row-major; mode 3: bias, fp32 row-major
__global__ void k_gemm(const __bf16* __restrict__ X, const __bf16* __restrict__ Wt,
                       const float* __restrict__ bias, void* __restrict__ outp,
                       int rows, int mode) {
  const int lane = threadIdx.x & 31;
  const int wave = threadIdx.x >> 5;
  const int h = lane >> 4, ln = lane & 15;
  const int task = blockIdx.x * 8 + wave;
  const int rb = (task >> 5) * 16;          // 512/16 = 32 col tiles
  const int cb = (task & 31) * 16;
  if (rb >= rows) return;

  v8f acc = {};
  const __bf16* arow = X + (size_t)(rb + ln) * 512;
  const __bf16* brow = Wt + (size_t)(cb + ln) * 512 + 16 * h;
  for (int k = 0; k < 512; k += 32) {
    v16bf a = load_a16(arow + k, h);
    v16bf b = load_b16(brow + k);
    acc = wmma_bf16(a, b, acc);
  }
  float bv = (mode >= 2) ? bias[cb + ln] : 0.0f;
  for (int v = 0; v < 8; ++v) {
    int r = rb + v + 8 * h;
    int c = cb + ln;
    float val = acc[v];
    if (mode == 0) {
      ((__bf16*)outp)[(size_t)r * 512 + c] = (__bf16)val;
    } else if (mode == 1) {
      ((__bf16*)outp)[(size_t)c * rows + r] = (__bf16)val;
    } else if (mode == 2) {
      val += bv; val = val > 0.0f ? val : 0.0f;
      ((__bf16*)outp)[(size_t)r * 512 + c] = (__bf16)val;
    } else {
      ((float*)outp)[(size_t)r * 512 + c] = val + bv;
    }
  }
}

// ---------------- flash attention ----------------
// block = 8 waves; wave task: (16-row query block) x (256-col half of D).
// All 8 waves share the same key stream -> K tile staged in LDS per jb.
__global__ void k_attn(const __bf16* __restrict__ Qb, const __bf16* __restrict__ Kb,
                       const __bf16* __restrict__ Vt, const float* __restrict__ Er,
                       float* __restrict__ X1) {
  __shared__ __bf16 sK[32][520];     // 32 keys x 512 kd, row padded (260 banks = 4 mod 64)
  __shared__ float  sP[8][16][32];   // per-wave P transpose staging
  const int tid  = threadIdx.x;
  const int lane = tid & 31;
  const int wave = tid >> 5;
  const int h = lane >> 4, ln = lane & 15;
  const int task = blockIdx.x * 8 + wave;
  const int qb = task >> 1;
  const int dh = task & 1;

  v8f z = {};
  v8f acc[16];
  float m[8], l[8];
  for (int c = 0; c < 16; ++c) acc[c] = z;
  for (int v = 0; v < 8; ++v) { m[v] = -__builtin_inff(); l[v] = 0.0f; }

  const __bf16* qrow = Qb + (size_t)(qb * 16 + ln) * 512;
  const size_t erow[1] = {0};
  (void)erow;

  for (int jb = 0; jb < 256; ++jb) {
    const int j0 = jb * 32;
    // ---- cooperative K-tile stage: 32 rows x 512 bf16, 16B per thread-chunk ----
    __syncthreads();  // previous iteration's readers are done
    for (int idx = tid; idx < 32 * 64; idx += 256) {
      int r = idx >> 6, c8 = idx & 63;
      *(uint4*)&sK[r][c8 * 8] =
          *(const uint4*)(Kb + (size_t)(j0 + r) * 512 + c8 * 8);
    }
    __syncthreads();  // staged tile visible to all waves

    // ---- S = Q Kt for 32 keys (two 16x16 tiles), fp32 accum ----
    v8f s0 = z, s1 = z;
    const __bf16* k0 = &sK[ln][16 * h];
    const __bf16* k1 = &sK[16 + ln][16 * h];
    for (int k = 0; k < 512; k += 32) {
      v16bf a = load_a16(qrow + k, h);
      s0 = wmma_bf16(a, load_b16(k0 + k), s0);
      s1 = wmma_bf16(a, load_b16(k1 + k), s1);
    }
    // ---- fused scale*mask (precomputed), online softmax ----
    float p0[8], p1[8], t[8];
    for (int v = 0; v < 8; ++v) {
      int i = qb * 16 + v + 8 * h;
      size_t ii = (size_t)(i & NMASK) * 4096;
      p0[v] = s0[v] * Er[ii + ((j0 + ln) & NMASK)];
      p1[v] = s1[v] * Er[ii + ((j0 + 16 + ln) & NMASK)];
      t[v] = fmaxf(p0[v], p1[v]);
    }
    for (int off = 8; off; off >>= 1)
      for (int v = 0; v < 8; ++v)
        t[v] = fmaxf(t[v], __shfl_xor(t[v], off, 32));
    float alpha[8], rs[8];
    for (int v = 0; v < 8; ++v) {
      float mn = fmaxf(m[v], t[v]);
      alpha[v] = __expf(m[v] - mn);
      m[v] = mn;
      p0[v] = __expf(p0[v] - mn);
      p1[v] = __expf(p1[v] - mn);
      rs[v] = p0[v] + p1[v];
    }
    for (int off = 8; off; off >>= 1)
      for (int v = 0; v < 8; ++v)
        rs[v] += __shfl_xor(rs[v], off, 32);
    for (int v = 0; v < 8; ++v) l[v] = l[v] * alpha[v] + rs[v];
    for (int c = 0; c < 16; ++c)
      for (int v = 0; v < 8; ++v)
        acc[c][v] *= alpha[v];
    // ---- transpose P (C-layout) -> A-layout via per-wave LDS tile ----
    for (int v = 0; v < 8; ++v) {
      sP[wave][v + 8 * h][ln] = p0[v];
      sP[wave][v + 8 * h][16 + ln] = p1[v];
    }
    asm volatile("s_wait_dscnt 0" ::: "memory");  // same-wave DS RAW
    V16U pa;
    for (int i2 = 0; i2 < 16; ++i2) {
      int kk = i2 + 16 * h;
      pa.e[i2] = (__bf16)sP[wave][ln][kk];
    }
    // ---- O += P @ V over this wave's 256-column D slice (Vt contiguous) ----
    const __bf16* vbase = Vt + (size_t)(dh * 256 + ln) * 8192 + j0 + 16 * h;
    for (int c = 0; c < 16; ++c) {
      v16bf b = load_b16(vbase + (size_t)c * 16 * 8192);
      acc[c] = wmma_bf16(pa.v, b, acc[c]);
    }
  }
  // ---- normalize and write fp32 ----
  for (int c = 0; c < 16; ++c)
    for (int v = 0; v < 8; ++v) {
      int r = qb * 16 + v + 8 * h;
      int d = dh * 256 + c * 16 + ln;
      X1[(size_t)r * 512 + d] = acc[c][v] / l[v];
    }
}

// ---------------- residual + layernorm (one block per row, 256 threads) ----------------
// useRL=1: x = A + Fcat(R,L); useRL=0: x = A + Badd. Writes fp32 and optional bf16.
__global__ void k_ln(const float* __restrict__ A, const float* __restrict__ Badd,
                     const float* __restrict__ R, const float* __restrict__ L,
                     const float* __restrict__ g, const float* __restrict__ bb,
                     float* __restrict__ outF, __bf16* __restrict__ outB, int useRL) {
  __shared__ float red[8];
  const int row = blockIdx.x;
  const int t = threadIdx.x;
  const float* res = useRL
      ? (row < 4096 ? R + (size_t)row * 512 : L + (size_t)(row - 4096) * 512)
      : Badd + (size_t)row * 512;
  float x0 = A[(size_t)row * 512 + t] + res[t];
  float x1 = A[(size_t)row * 512 + t + 256] + res[t + 256];

  float s = x0 + x1;
  for (int off = 16; off; off >>= 1) s += __shfl_xor(s, off, 32);
  if ((t & 31) == 0) red[t >> 5] = s;
  __syncthreads();
  float tot = 0.0f;
  for (int i = 0; i < 8; ++i) tot += red[i];
  float mu = tot * (1.0f / 512.0f);
  __syncthreads();

  float d0 = x0 - mu, d1 = x1 - mu;
  float v2 = d0 * d0 + d1 * d1;
  for (int off = 16; off; off >>= 1) v2 += __shfl_xor(v2, off, 32);
  if ((t & 31) == 0) red[t >> 5] = v2;
  __syncthreads();
  float var = 0.0f;
  for (int i = 0; i < 8; ++i) var += red[i];
  var *= (1.0f / 512.0f);
  float rstd = rsqrtf(var + 1e-5f);

  float y0 = d0 * rstd * g[t] + bb[t];
  float y1 = d1 * rstd * g[t + 256] + bb[t + 256];
  outF[(size_t)row * 512 + t] = y0;
  outF[(size_t)row * 512 + t + 256] = y1;
  if (outB) {
    outB[(size_t)row * 512 + t] = (__bf16)y0;
    outB[(size_t)row * 512 + t + 256] = (__bf16)y1;
  }
}

extern "C" void kernel_launch(void* const* d_in, const int* in_sizes, int n_in,
                              void* d_out, int out_size, void* d_ws, size_t ws_size,
                              hipStream_t stream) {
  const float* R    = (const float*)d_in[0];
  const float* L    = (const float*)d_in[1];
  /* d_in[2] = Dg (unused by reference) */
  const float* De   = (const float*)d_in[3];
  const int*   Dist = (const int*)d_in[4];
  const float* WQ   = (const float*)d_in[5];
  const float* WK   = (const float*)d_in[6];
  const float* WV   = (const float*)d_in[7];
  const float* g1   = (const float*)d_in[8];
  const float* bln1 = (const float*)d_in[9];
  const float* W1   = (const float*)d_in[10];
  const float* b1   = (const float*)d_in[11];
  const float* W2   = (const float*)d_in[12];
  const float* b2   = (const float*)d_in[13];
  const float* g2   = (const float*)d_in[14];
  const float* bln2 = (const float*)d_in[15];

  char* w = (char*)d_ws;
  auto alloc = [&](size_t bytes) {
    void* p = (void*)w;
    w += (bytes + 255) & ~(size_t)255;
    return p;
  };
  const size_t nd = (size_t)TWO_N * DD;
  __bf16* Fcat = (__bf16*)alloc(nd * 2);
  __bf16* Qb   = (__bf16*)alloc(nd * 2);
  __bf16* Kb   = (__bf16*)alloc(nd * 2);
  __bf16* Vtb  = (__bf16*)alloc(nd * 2);
  __bf16* Wqt  = (__bf16*)alloc(512 * 512 * 2);
  __bf16* Wkt  = (__bf16*)alloc(512 * 512 * 2);
  __bf16* Wvt  = (__bf16*)alloc(512 * 512 * 2);
  __bf16* W1t  = (__bf16*)alloc(512 * 512 * 2);
  __bf16* W2t  = (__bf16*)alloc(512 * 512 * 2);
  float*  X1   = (float*)alloc(nd * 4);
  float*  Fp   = (float*)alloc(nd * 4);
  __bf16* Fpb  = (__bf16*)alloc(nd * 2);
  __bf16* Hb   = (__bf16*)alloc(nd * 2);
  float*  Y    = (float*)alloc(nd * 4);
  float*  Er   = (float*)alloc((size_t)4096 * 4096 * 4);

  // prep
  k_fcat<<<(int)(nd / 256), 256, 0, stream>>>(R, L, Fcat);
  k_wt<<<1024, 256, 0, stream>>>(WQ, Wqt);
  k_wt<<<1024, 256, 0, stream>>>(WK, Wkt);
  k_wt<<<1024, 256, 0, stream>>>(WV, Wvt);
  k_wt<<<1024, 256, 0, stream>>>(W1, W1t);
  k_wt<<<1024, 256, 0, stream>>>(W2, W2t);
  k_er<<<65536, 256, 0, stream>>>(De, Dist, Er);

  // QKV projections (16384 tiles / 8 waves = 2048 blocks)
  k_gemm<<<2048, 256, 0, stream>>>(Fcat, Wqt, nullptr, Qb,  TWO_N, 0);
  k_gemm<<<2048, 256, 0, stream>>>(Fcat, Wkt, nullptr, Kb,  TWO_N, 0);
  k_gemm<<<2048, 256, 0, stream>>>(Fcat, Wvt, nullptr, Vtb, TWO_N, 1);

  // flash attention: 512 query blocks * 2 D-halves = 1024 wave tasks
  k_attn<<<128, 256, 0, stream>>>(Qb, Kb, Vtb, Er, X1);

  // LN1 (residual from Fcat), produce fp32 + bf16
  k_ln<<<TWO_N, 256, 0, stream>>>(X1, nullptr, R, L, g1, bln1, Fp, Fpb, 1);

  // MLP
  k_gemm<<<2048, 256, 0, stream>>>(Fpb, W1t, b1, Hb, TWO_N, 2);
  k_gemm<<<2048, 256, 0, stream>>>(Hb,  W2t, b2, Y,  TWO_N, 3);

  // LN2 -> output
  k_ln<<<TWO_N, 256, 0, stream>>>(Fp, Y, nullptr, nullptr, g2, bln2,
                                  (float*)d_out, nullptr, 0);
}
